// HyperGLayer_12695923327692
// MI455X (gfx1250) — matrix operand order repaired and checked by reading, not compiled, not run on hardware
//
#include <hip/hip_runtime.h>
#include <hip/hip_bf16.h>

// ---------------- problem constants (match reference) ----------------
#define H_N   100000
#define E_N   200000
#define NI_N  1000000
#define D_H   128
#define D_E   64
#define D_OUT 128
#define LN_EPS 1e-5f

// ---------------- vector types ----------------
typedef _Float16 v16h __attribute__((ext_vector_type(16)));
typedef _Float16 v8h  __attribute__((ext_vector_type(8)));
typedef _Float16 v4h  __attribute__((ext_vector_type(4)));
typedef float    v8f  __attribute__((ext_vector_type(8)));
typedef float    f32x4 __attribute__((ext_vector_type(4)));

// ---------------- workspace layout (bytes) ----------------
// all offsets multiples of 128
static const size_t OFF_ESUM = 0;                         // E*128 f32 = 102,400,000
static const size_t OFF_ECNT = 102400000;                 // E f32    =     800,000
static const size_t OFF_NSUM = 103200000;                 // H*128 f32=  51,200,000
static const size_t OFF_NCNT = 154400000;                 // H f32    =     400,000
static const size_t OFF_ZERO_END = 154800000;             // everything above is zeroed each launch
static const size_t OFF_H    = 154800000;                 // E*128 f16=  51,200,000
static const size_t OFF_EMSG = 206000000;                 // E*128 f16=  51,200,000
static const size_t OFF_T    = 257200000;                 // H*128 f16=  25,600,000
static const size_t OFF_PE1  = 282800000;                 // 6*8*512 f16 = 49,152
static const size_t OFF_PE2  = OFF_PE1 + 49152;           // 4*8*512 f16 = 32,768
static const size_t OFF_PN1  = OFF_PE2 + 32768;
static const size_t OFF_PN2  = OFF_PN1 + 32768;
static const size_t OFF_PP   = OFF_PN2 + 32768;

// =====================================================================
// zero the accumulator regions
__global__ __launch_bounds__(256) void k_zero(f32x4* p, int n4) {
  int stride = gridDim.x * blockDim.x;
  for (int i = blockIdx.x * blockDim.x + threadIdx.x; i < n4; i += stride) {
    f32x4 z = {0.f, 0.f, 0.f, 0.f};
    p[i] = z;
  }
}

// =====================================================================
// pack W [128][Kdim] (torch Linear [out,in]) into WMMA B-fragment-major f16.
// B is K x N with B[k][n] = W[n][k].  B fragment (ISA 7.12.2, 16-bit, wave32):
//   lane L, half-element e:  K = kt*32 + (L>>4)*8 + (e<8 ? e : 8+e),  N = nt*16 + (L&15)
// fragment f = kt*8+nt stored as 512 contiguous halves: [lane][e]
__global__ __launch_bounds__(512) void k_pack(const float* __restrict__ W, int Kdim,
                                              _Float16* __restrict__ packed) {
  int f    = blockIdx.x;          // kt*8 + nt
  int kt   = f >> 3, nt = f & 7;
  int lane = threadIdx.x >> 4;    // 0..31
  int e    = threadIdx.x & 15;    // 0..15
  int K = kt * 32 + (lane >> 4) * 8 + ((e < 8) ? e : 8 + e);
  int N = nt * 16 + (lane & 15);
  packed[(size_t)f * 512 + lane * 16 + e] = (_Float16)W[(size_t)N * Kdim + K];
}

// =====================================================================
// scatter 1: edge_sum[res[i]] += x_h[fg[i]] ; edge_cnt[res[i]] += 1
// one wave per incidence; 32 lanes cover 128 channels (float4 each)
__global__ __launch_bounds__(256) void k_scatter1(const float* __restrict__ xh,
                                                  const int* __restrict__ fg,
                                                  const int* __restrict__ res,
                                                  float* __restrict__ esum,
                                                  float* __restrict__ ecnt) {
  int wid  = (blockIdx.x * blockDim.x + threadIdx.x) >> 5;
  int lane = threadIdx.x & 31;
  if (wid >= NI_N) return;
  int src = fg[wid];
  int dst = res[wid];
  f32x4 v = *(const f32x4*)(xh + (size_t)src * 128 + lane * 4);
  float* er = esum + (size_t)dst * 128 + lane * 4;
  atomicAdd(er + 0, v[0]); atomicAdd(er + 1, v[1]);
  atomicAdd(er + 2, v[2]); atomicAdd(er + 3, v[3]);
  if (lane == 0) atomicAdd(ecnt + dst, 1.0f);
}

// scatter 2: node_sum[fg[i]] += edge_msg[res[i]] (f16 src) ; node_cnt += 1
__global__ __launch_bounds__(256) void k_scatter2(const _Float16* __restrict__ emsg,
                                                  const int* __restrict__ fg,
                                                  const int* __restrict__ res,
                                                  float* __restrict__ nsum,
                                                  float* __restrict__ ncnt) {
  int wid  = (blockIdx.x * blockDim.x + threadIdx.x) >> 5;
  int lane = threadIdx.x & 31;
  if (wid >= NI_N) return;
  int e = res[wid];
  int n = fg[wid];
  v4h m = *(const v4h*)(emsg + (size_t)e * 128 + lane * 4);
  float* nr = nsum + (size_t)n * 128 + lane * 4;
  atomicAdd(nr + 0, (float)m[0]); atomicAdd(nr + 1, (float)m[1]);
  atomicAdd(nr + 2, (float)m[2]); atomicAdd(nr + 3, (float)m[3]);
  if (lane == 0) atomicAdd(ncnt + n, 1.0f);
}

// =====================================================================
// WMMA helper
__device__ __forceinline__ v8f wmma_f16(v16h a, v16h b, v8f c) {
  return __builtin_amdgcn_wmma_f32_16x16x32_f16(false, a, false, b, (short)0, c,
                                                false, false);
}

// =====================================================================
// edge MLP layer 1: h = relu( [edge_mean | attr] @ We1^T + be1 ) -> f16 [E][128]
// one wave per 16-edge tile; A built per ISA 16-bit A layout:
//   lane L: row M = L&15 ; halves e<8 -> K = kt*32 + hi*8 + e ; e>=8 -> +16
__global__ __launch_bounds__(256) void k_edge_mlp1(const float* __restrict__ esum,
                                                   const float* __restrict__ ecnt,
                                                   const float* __restrict__ attr,
                                                   const _Float16* __restrict__ pB,
                                                   const float* __restrict__ be1,
                                                   _Float16* __restrict__ hbuf,
                                                   int ntiles) {
  __shared__ _Float16 stage[8][16][128];
  int wave = threadIdx.x >> 5, lane = threadIdx.x & 31;
  int n16 = lane & 15, hi = lane >> 4;
  for (int tile = blockIdx.x * 8 + wave; tile < ntiles; tile += gridDim.x * 8) {
    int row = tile * 16 + n16;
    float ic = 1.0f / fmaxf(ecnt[row], 1.0f);
    v8f acc[8] = {};
    #pragma unroll
    for (int kt = 0; kt < 6; ++kt) {
      v16h a;
      int kbase = kt * 32 + hi * 8;
      #pragma unroll
      for (int rsel = 0; rsel < 2; ++rsel) {
        int k0 = kbase + rsel * 16;
        f32x4 u, v; float s;
        if (kt < 4) {  // K in [0,128): node-mean part
          const float* p = esum + (size_t)row * 128 + k0;
          u = *(const f32x4*)p; v = *(const f32x4*)(p + 4); s = ic;
        } else {       // K in [128,192): edge attr part
          const float* p = attr + (size_t)row * 64 + (k0 - 128);
          u = *(const f32x4*)p; v = *(const f32x4*)(p + 4); s = 1.0f;
        }
        #pragma unroll
        for (int e = 0; e < 4; ++e) {
          a[rsel * 8 + e]     = (_Float16)(u[e] * s);
          a[rsel * 8 + 4 + e] = (_Float16)(v[e] * s);
        }
      }
      #pragma unroll
      for (int nt = 0; nt < 8; ++nt) {
        v16h b = ((const v16h*)pB)[(kt * 8 + nt) * 32 + lane];
        acc[nt] = wmma_f16(a, b, acc[nt]);
      }
    }
    // bias + relu, stage through per-wave LDS (D layout -> row major)
    #pragma unroll
    for (int nt = 0; nt < 8; ++nt) {
      int col = nt * 16 + n16;
      float b = be1[col];
      #pragma unroll
      for (int r = 0; r < 8; ++r)
        stage[wave][r + hi * 8][col] = (_Float16)fmaxf(acc[nt][r] + b, 0.0f);
    }
    #pragma unroll
    for (int rr = 0; rr < 16; ++rr)
      *(v4h*)(hbuf + ((size_t)tile * 16 + rr) * 128 + lane * 4) =
          *(const v4h*)&stage[wave][rr][lane * 4];
  }
}

// edge MLP layer 2: edge_msg = h @ We2^T + be2 -> f16 [E][128]
__global__ __launch_bounds__(256) void k_edge_mlp2(const _Float16* __restrict__ hbuf,
                                                   const _Float16* __restrict__ pB,
                                                   const float* __restrict__ be2,
                                                   _Float16* __restrict__ emsg,
                                                   int ntiles) {
  __shared__ _Float16 stage[8][16][128];
  int wave = threadIdx.x >> 5, lane = threadIdx.x & 31;
  int n16 = lane & 15, hi = lane >> 4;
  for (int tile = blockIdx.x * 8 + wave; tile < ntiles; tile += gridDim.x * 8) {
    int row = tile * 16 + n16;
    v8f acc[8] = {};
    #pragma unroll
    for (int kt = 0; kt < 4; ++kt) {
      int kbase = kt * 32 + hi * 8;
      v8h r1 = *(const v8h*)(hbuf + (size_t)row * 128 + kbase);
      v8h r2 = *(const v8h*)(hbuf + (size_t)row * 128 + kbase + 16);
      v16h a = __builtin_shufflevector(r1, r2, 0, 1, 2, 3, 4, 5, 6, 7,
                                       8, 9, 10, 11, 12, 13, 14, 15);
      #pragma unroll
      for (int nt = 0; nt < 8; ++nt) {
        v16h b = ((const v16h*)pB)[(kt * 8 + nt) * 32 + lane];
        acc[nt] = wmma_f16(a, b, acc[nt]);
      }
    }
    #pragma unroll
    for (int nt = 0; nt < 8; ++nt) {
      int col = nt * 16 + n16;
      float b = be2[col];
      #pragma unroll
      for (int r = 0; r < 8; ++r)
        stage[wave][r + hi * 8][col] = (_Float16)(acc[nt][r] + b);
    }
    #pragma unroll
    for (int rr = 0; rr < 16; ++rr)
      *(v4h*)(emsg + ((size_t)tile * 16 + rr) * 128 + lane * 4) =
          *(const v4h*)&stage[wave][rr][lane * 4];
  }
}

// node MLP layer 1: t = relu( node_mean @ Wn1^T + bn1 ) -> f16 [H][128]
__global__ __launch_bounds__(256) void k_node_mlp1(const float* __restrict__ nsum,
                                                   const float* __restrict__ ncnt,
                                                   const _Float16* __restrict__ pB,
                                                   const float* __restrict__ bn1,
                                                   _Float16* __restrict__ tbuf,
                                                   int ntiles) {
  __shared__ _Float16 stage[8][16][128];
  int wave = threadIdx.x >> 5, lane = threadIdx.x & 31;
  int n16 = lane & 15, hi = lane >> 4;
  for (int tile = blockIdx.x * 8 + wave; tile < ntiles; tile += gridDim.x * 8) {
    int row = tile * 16 + n16;
    float ic = 1.0f / fmaxf(ncnt[row], 1.0f);
    v8f acc[8] = {};
    #pragma unroll
    for (int kt = 0; kt < 4; ++kt) {
      v16h a;
      int kbase = kt * 32 + hi * 8;
      #pragma unroll
      for (int rsel = 0; rsel < 2; ++rsel) {
        int k0 = kbase + rsel * 16;
        const float* p = nsum + (size_t)row * 128 + k0;
        f32x4 u = *(const f32x4*)p, v = *(const f32x4*)(p + 4);
        #pragma unroll
        for (int e = 0; e < 4; ++e) {
          a[rsel * 8 + e]     = (_Float16)(u[e] * ic);
          a[rsel * 8 + 4 + e] = (_Float16)(v[e] * ic);
        }
      }
      #pragma unroll
      for (int nt = 0; nt < 8; ++nt) {
        v16h b = ((const v16h*)pB)[(kt * 8 + nt) * 32 + lane];
        acc[nt] = wmma_f16(a, b, acc[nt]);
      }
    }
    #pragma unroll
    for (int nt = 0; nt < 8; ++nt) {
      int col = nt * 16 + n16;
      float b = bn1[col];
      #pragma unroll
      for (int r = 0; r < 8; ++r)
        stage[wave][r + hi * 8][col] = (_Float16)fmaxf(acc[nt][r] + b, 0.0f);
    }
    #pragma unroll
    for (int rr = 0; rr < 16; ++rr)
      *(v4h*)(tbuf + ((size_t)tile * 16 + rr) * 128 + lane * 4) =
          *(const v4h*)&stage[wave][rr][lane * 4];
  }
}

// node MLP layer 2 fused with projection + relu + layernorm:
//   z = relu( x_h@Wp^T + bp + t@Wn2^T + bn2 ); out = LN(z)*gamma + beta
__global__ __launch_bounds__(128) void k_node_mlp2_ln(const float* __restrict__ xh,
                                                      const _Float16* __restrict__ tbuf,
                                                      const _Float16* __restrict__ pBp,
                                                      const _Float16* __restrict__ pBn2,
                                                      const float* __restrict__ bp,
                                                      const float* __restrict__ bn2,
                                                      const float* __restrict__ gamma,
                                                      const float* __restrict__ beta,
                                                      float* __restrict__ out,
                                                      int ntiles) {
  __shared__ float stage[4][16][128];
  int wave = threadIdx.x >> 5, lane = threadIdx.x & 31;
  int n16 = lane & 15, hi = lane >> 4;
  for (int tile = blockIdx.x * 4 + wave; tile < ntiles; tile += gridDim.x * 4) {
    int row = tile * 16 + n16;
    v8f acc[8] = {};
    #pragma unroll
    for (int kt = 0; kt < 4; ++kt) {
      int kbase = kt * 32 + hi * 8;
      // A from x_h (f32 -> f16)
      v16h ax;
      #pragma unroll
      for (int rsel = 0; rsel < 2; ++rsel) {
        int k0 = kbase + rsel * 16;
        const float* p = xh + (size_t)row * 128 + k0;
        f32x4 u = *(const f32x4*)p, v = *(const f32x4*)(p + 4);
        #pragma unroll
        for (int e = 0; e < 4; ++e) {
          ax[rsel * 8 + e]     = (_Float16)u[e];
          ax[rsel * 8 + 4 + e] = (_Float16)v[e];
        }
      }
      // A from t (f16)
      v8h r1 = *(const v8h*)(tbuf + (size_t)row * 128 + kbase);
      v8h r2 = *(const v8h*)(tbuf + (size_t)row * 128 + kbase + 16);
      v16h at = __builtin_shufflevector(r1, r2, 0, 1, 2, 3, 4, 5, 6, 7,
                                        8, 9, 10, 11, 12, 13, 14, 15);
      #pragma unroll
      for (int nt = 0; nt < 8; ++nt) {
        v16h b1 = ((const v16h*)pBp)[(kt * 8 + nt) * 32 + lane];
        acc[nt] = wmma_f16(ax, b1, acc[nt]);
        v16h b2 = ((const v16h*)pBn2)[(kt * 8 + nt) * 32 + lane];
        acc[nt] = wmma_f16(at, b2, acc[nt]);
      }
    }
    // bias + relu (in place), then row statistics
    float s1[8], s2[8];
    #pragma unroll
    for (int r = 0; r < 8; ++r) { s1[r] = 0.f; s2[r] = 0.f; }
    #pragma unroll
    for (int nt = 0; nt < 8; ++nt) {
      int col = nt * 16 + n16;
      float b = bp[col] + bn2[col];
      #pragma unroll
      for (int r = 0; r < 8; ++r) {
        float z = fmaxf(acc[nt][r] + b, 0.0f);
        acc[nt][r] = z;
        s1[r] += z;
        s2[r] += z * z;
      }
    }
    // butterfly over the 16 lanes holding a row (xor masks stay within half)
    #pragma unroll
    for (int off = 1; off < 16; off <<= 1) {
      #pragma unroll
      for (int r = 0; r < 8; ++r) {
        s1[r] += __shfl_xor(s1[r], off, 32);
        s2[r] += __shfl_xor(s2[r], off, 32);
      }
    }
    float mu[8], rs[8];
    #pragma unroll
    for (int r = 0; r < 8; ++r) {
      mu[r] = s1[r] * (1.0f / 128.0f);
      float var = s2[r] * (1.0f / 128.0f) - mu[r] * mu[r];
      rs[r] = rsqrtf(var + LN_EPS);
    }
    #pragma unroll
    for (int nt = 0; nt < 8; ++nt) {
      int col = nt * 16 + n16;
      float g = gamma[col], bb = beta[col];
      #pragma unroll
      for (int r = 0; r < 8; ++r)
        stage[wave][r + hi * 8][col] = (acc[nt][r] - mu[r]) * rs[r] * g + bb;
    }
    #pragma unroll
    for (int rr = 0; rr < 16; ++rr)
      *(f32x4*)(out + ((size_t)tile * 16 + rr) * 128 + lane * 4) =
          *(const f32x4*)&stage[wave][rr][lane * 4];
  }
}

// =====================================================================
extern "C" void kernel_launch(void* const* d_in, const int* in_sizes, int n_in,
                              void* d_out, int out_size, void* d_ws, size_t ws_size,
                              hipStream_t stream) {
  (void)in_sizes; (void)n_in; (void)out_size; (void)ws_size;
  const float* xh    = (const float*)d_in[0];
  const float* attr  = (const float*)d_in[1];
  const float* Wp    = (const float*)d_in[2];
  const float* bp    = (const float*)d_in[3];
  const float* We1   = (const float*)d_in[4];
  const float* be1   = (const float*)d_in[5];
  const float* We2   = (const float*)d_in[6];
  const float* be2   = (const float*)d_in[7];
  const float* Wn1   = (const float*)d_in[8];
  const float* bn1   = (const float*)d_in[9];
  const float* Wn2   = (const float*)d_in[10];
  const float* bn2   = (const float*)d_in[11];
  const float* gamma = (const float*)d_in[12];
  const float* beta  = (const float*)d_in[13];
  const int*   fg    = (const int*)d_in[14];
  const int*   res   = (const int*)d_in[15];

  char* w = (char*)d_ws;
  float*    esum = (float*)(w + OFF_ESUM);
  float*    ecnt = (float*)(w + OFF_ECNT);
  float*    nsum = (float*)(w + OFF_NSUM);
  float*    ncnt = (float*)(w + OFF_NCNT);
  _Float16* hbuf = (_Float16*)(w + OFF_H);
  _Float16* emsg = (_Float16*)(w + OFF_EMSG);
  _Float16* tbuf = (_Float16*)(w + OFF_T);
  _Float16* pe1  = (_Float16*)(w + OFF_PE1);
  _Float16* pe2  = (_Float16*)(w + OFF_PE2);
  _Float16* pn1  = (_Float16*)(w + OFF_PN1);
  _Float16* pn2  = (_Float16*)(w + OFF_PN2);
  _Float16* pp   = (_Float16*)(w + OFF_PP);

  // 1) zero accumulators (must happen every launch)
  int n4 = (int)(OFF_ZERO_END / 16);
  k_zero<<<2048, 256, 0, stream>>>((f32x4*)w, n4);

  // 2) pack weights into WMMA B-fragment layout (hot in L2 afterwards)
  k_pack<<<48, 512, 0, stream>>>(We1, 192, pe1);
  k_pack<<<32, 512, 0, stream>>>(We2, 128, pe2);
  k_pack<<<32, 512, 0, stream>>>(Wn1, 128, pn1);
  k_pack<<<32, 512, 0, stream>>>(Wn2, 128, pn2);
  k_pack<<<32, 512, 0, stream>>>(Wp, 128, pp);

  // 3) node -> edge scatter-sum (+counts); one wave per incidence
  k_scatter1<<<NI_N / 8, 256, 0, stream>>>(xh, fg, res, esum, ecnt);

  // 4) edge MLP (WMMA f16, f32 accumulate)
  int etiles = E_N / 16;  // 12500
  k_edge_mlp1<<<(etiles + 7) / 8, 256, 0, stream>>>(esum, ecnt, attr, pe1, be1,
                                                    hbuf, etiles);
  k_edge_mlp2<<<(etiles + 7) / 8, 256, 0, stream>>>(hbuf, pe2, be2, emsg, etiles);

  // 5) edge -> node scatter-sum (+counts); f16 messages halve gather traffic
  k_scatter2<<<NI_N / 8, 256, 0, stream>>>(emsg, fg, res, nsum, ncnt);

  // 6) node MLPs + projection + relu + layernorm (WMMA)
  int ntiles = H_N / 16;  // 6250
  k_node_mlp1<<<(ntiles + 7) / 8, 256, 0, stream>>>(nsum, ncnt, pn1, bn1, tbuf,
                                                    ntiles);
  k_node_mlp2_ln<<<(ntiles + 3) / 4, 128, 0, stream>>>(xh, tbuf, pp, pn2, bp, bn2,
                                                       gamma, beta, (float*)d_out,
                                                       ntiles);
}
